// Hecto_23991687315787
// MI455X (gfx1250) — matrix operand order repaired and checked by reading, not compiled, not run on hardware
//
#include <hip/hip_runtime.h>

// ---------------- types / helpers ----------------
typedef __attribute__((ext_vector_type(16))) __bf16 bf16x16;
typedef __attribute__((ext_vector_type(8)))  float  f32x8;

union Frag16 { uint4 q[2]; bf16x16 v; };

__device__ __forceinline__ unsigned short f2bf(float f) {
  union { float f; unsigned u; } v; v.f = f;
  unsigned r = (v.u + 0x7FFFu + ((v.u >> 16) & 1u)) >> 16;
  return (unsigned short)r;
}

// constants
#define NB 8
#define NTOK 1360          // 1024+256+64+16 tokens per batch
#define CIN 256
#define COUT 768
#define NTILES 85          // 64+16+4+1 m-tiles per batch
#define THRL (-1.3862943611198906f)   // logit(0.2)

// decode m-tile -> (scale, batch-local token base, scale-local pixel base)
__device__ __forceinline__ void decode_tile(int mt, int& s, int& ltok, int& pbase) {
  if (mt < 64)       { s = 0; ltok = mt * 16;               pbase = ltok; }
  else if (mt < 80)  { s = 1; ltok = 1024 + (mt - 64) * 16; pbase = ltok - 1024; }
  else if (mt < 84)  { s = 2; ltok = 1280 + (mt - 80) * 16; pbase = ltok - 1280; }
  else               { s = 3; ltok = 1344;                  pbase = 0; }
}

// ---------------- KA: fused (feat+pos)*(1-mask) + 2x2 avgpool -> bf16 tokens ----------------
__global__ __launch_bounds__(256) void kA_pool(
    const float* __restrict__ f0, const float* __restrict__ f1,
    const float* __restrict__ f2, const float* __restrict__ f3,
    const float* __restrict__ p0, const float* __restrict__ p1,
    const float* __restrict__ p2, const float* __restrict__ p3,
    const unsigned char* __restrict__ m0, const unsigned char* __restrict__ m1,
    const unsigned char* __restrict__ m2, const unsigned char* __restrict__ m3,
    unsigned short* __restrict__ pooled, float* __restrict__ mpinv) {
  const int tid = threadIdx.x;
  const int b  = blockIdx.x / NTILES;
  const int mt = blockIdx.x % NTILES;
  int s, ltok, pbase; decode_tile(mt, s, ltok, pbase);
  const int Hs[4] = {64, 32, 16, 8};
  const int H = Hs[s], PO = H >> 1;
  const float* F; const float* P; const unsigned char* M;
  switch (s) {
    case 0: F = f0; P = p0; M = m0; break;
    case 1: F = f1; P = p1; M = m1; break;
    case 2: F = f2; P = p2; M = m2; break;
    default: F = f3; P = p3; M = m3; break;
  }
  const int j  = tid & 15;   // pixel within tile
  const int c0 = tid >> 4;   // channel slice
  const int pidx = pbase + j;
  const int py = pidx / PO, px = pidx % PO;
  const int y = 2 * py, x = 2 * px;

  const unsigned char* mb = M + ((size_t)b * H + y) * H + x;
  const float w00 = mb[0]     ? 0.f : 1.f;
  const float w01 = mb[1]     ? 0.f : 1.f;
  const float w10 = mb[H]     ? 0.f : 1.f;
  const float w11 = mb[H + 1] ? 0.f : 1.f;
  if (c0 == 0) {
    float cnt = 0.25f * (w00 + w01 + w10 + w11);
    mpinv[b * NTOK + ltok + j] = 1.0f / fmaxf(cnt, 1e-6f);
  }

  __shared__ __align__(16) unsigned short Wls[16 * 264];  // padded stride
  #pragma unroll
  for (int ci = 0; ci < 16; ++ci) {
    const int c = c0 + ci * 16;
    const size_t base = (((size_t)b * CIN + c) * H + y) * H + x;
    float2 fa = *(const float2*)(F + base);
    float2 fb = *(const float2*)(F + base + H);
    float2 pa = *(const float2*)(P + base);
    float2 pb = *(const float2*)(P + base + H);
    float sum = (fa.x + pa.x) * w00 + (fa.y + pa.y) * w01 +
                (fb.x + pb.x) * w10 + (fb.y + pb.y) * w11;
    Wls[j * 264 + c] = f2bf(0.25f * sum);
  }
  __syncthreads();
  const size_t gbase = ((size_t)(b * NTOK + ltok)) * CIN;
  #pragma unroll
  for (int i = 0; i < 16; ++i) {
    int e = tid + 256 * i;
    pooled[gbase + e] = Wls[(e >> 8) * 264 + (e & 255)];  // contiguous writes
  }
}

// ---------------- KB: convW (4x768x256) f32 -> bf16; zero ctx accumulators ----------------
__global__ __launch_bounds__(256) void kB_prep(
    const float* __restrict__ w0, const float* __restrict__ w1,
    const float* __restrict__ w2, const float* __restrict__ w3,
    unsigned short* __restrict__ wbf, float* __restrict__ ctxsum) {
  int i = blockIdx.x * 256 + threadIdx.x;
  if (i < NB * COUT) ctxsum[i] = 0.f;
  if (i < 4 * COUT * CIN) {
    int s = i / (COUT * CIN), r = i % (COUT * CIN);
    const float* src = (s == 0) ? w0 : (s == 1) ? w1 : (s == 2) ? w2 : w3;
    wbf[i] = f2bf(src[r]);
  }
}

// ---------------- KC: WMMA GEMM (16 tok x 768 out, K=256) + LN + batch accumulate ----------------
__global__ __launch_bounds__(256) void kC_gemm_ln(
    const unsigned short* __restrict__ pooled, const unsigned short* __restrict__ wbf,
    const float* __restrict__ mpinv_g,
    const float* __restrict__ cb0, const float* __restrict__ cb1,
    const float* __restrict__ cb2, const float* __restrict__ cb3,
    const float* __restrict__ g0, const float* __restrict__ g1,
    const float* __restrict__ g2, const float* __restrict__ g3,
    float* __restrict__ ctxsum) {
  const int tid = threadIdx.x;
  const int b  = blockIdx.x / NTILES;
  const int mt = blockIdx.x % NTILES;
  int s, ltok, pbase; decode_tile(mt, s, ltok, pbase);
  const float* convb = (s == 0) ? cb0 : (s == 1) ? cb1 : (s == 2) ? cb2 : cb3;
  const float* lng   = (s == 0) ? g0  : (s == 1) ? g1  : (s == 2) ? g2  : g3;

  __shared__ __align__(16) unsigned short As[16 * 264];  // 16 tokens x 256 (padded)
  __shared__ float Cls[16 * 772];                        // 16 tokens x 768 (padded)
  __shared__ float mpi[16], meanA[16], rstdA[16];

  // stage A-tile (16x256 bf16, contiguous in global) into LDS
  const unsigned short* gA = pooled + ((size_t)(b * NTOK + ltok)) * CIN;
  #pragma unroll
  for (int i = tid; i < 512; i += 256) {            // 512 chunks of 8 bf16
    int goff = i * 8, row = goff >> 8, c = goff & 255;
    *reinterpret_cast<uint4*>(&As[row * 264 + c]) =
        *reinterpret_cast<const uint4*>(gA + goff);
  }
  if (tid < 16) mpi[tid] = mpinv_g[b * NTOK + ltok + tid];
  __syncthreads();

  const int lane = tid & 31, w = tid >> 5;
  const int half = lane >> 4, l16 = lane & 15;
  const unsigned short* gW = wbf + (size_t)s * COUT * CIN;

  // hoist all A fragments (8 K-steps) into registers: reused across 6 n-tiles
  Frag16 afr[8];
  #pragma unroll
  for (int kk = 0; kk < 8; ++kk) {
    const int akb = kk * 32 + half * 8;
    const uint4* ap = reinterpret_cast<const uint4*>(&As[l16 * 264 + akb]);
    afr[kk].q[0] = ap[0];
    afr[kk].q[1] = ap[2];  // +16 bf16 elements
  }

  for (int nt = 0; nt < 6; ++nt) {
    const int colbase = w * 96 + nt * 16;
    f32x8 acc = {0.f, 0.f, 0.f, 0.f, 0.f, 0.f, 0.f, 0.f};
    const int n = colbase + l16;
    #pragma unroll
    for (int kk = 0; kk < 8; ++kk) {
      const int k0 = kk * 32;
      Frag16 bf;
      // B frag: col n; lane<16 -> K k0..k0+7, k0+16..k0+23; lane>=16 -> k0+8..k0+15, k0+24..k0+31
      const int bkb = k0 + half * 8;
      const uint4* bp = reinterpret_cast<const uint4*>(gW + (size_t)n * CIN + bkb);
      bf.q[0] = bp[0];
      bf.q[1] = bp[2];
      acc = __builtin_amdgcn_wmma_f32_16x16x32_bf16(
          false, afr[kk].v, false, bf.v, (short)0, acc, false, false);
    }
    const float cbv = convb[n];
    #pragma unroll
    for (int r = 0; r < 8; ++r) {
      const int row = r + half * 8;                 // D layout: VGPR r -> M=r(+8)
      Cls[row * 772 + n] = (acc[r] + cbv) * mpi[row];
    }
  }
  __syncthreads();

  // per-token LayerNorm stats (16 threads / token)
  const int t = tid >> 4, su = tid & 15;
  float s1 = 0.f, s2 = 0.f;
  #pragma unroll
  for (int jj = 0; jj < 48; ++jj) {
    float xv = Cls[t * 772 + su + jj * 16];
    s1 += xv; s2 += xv * xv;
  }
  #pragma unroll
  for (int m = 8; m; m >>= 1) {
    s1 += __shfl_xor(s1, m, 16);
    s2 += __shfl_xor(s2, m, 16);
  }
  const float mean = s1 * (1.f / 768.f);
  const float var  = s2 * (1.f / 768.f) - mean * mean;
  const float rstd = rsqrtf(var + 1e-5f);
  if (su == 0) { meanA[t] = mean; rstdA[t] = rstd; }
  __syncthreads();

  // column-wise sum over 16 tokens of g*(x-mean)*rstd -> batch accumulator
  for (int col = tid; col < COUT; col += 256) {
    float a2 = 0.f;
    #pragma unroll
    for (int t2 = 0; t2 < 16; ++t2)
      a2 += (Cls[t2 * 772 + col] - meanA[t2]) * rstdA[t2];
    atomicAdd(&ctxsum[b * COUT + col], lng[col] * a2);
  }
}

// ---------------- KD: prompt pooling + projection (per batch) ----------------
__global__ __launch_bounds__(256) void kD_prompt(
    const float* __restrict__ logits, const float* __restrict__ hs,
    const float* __restrict__ Wq, const float* __restrict__ bq,
    float* __restrict__ prompt) {
  const int b = blockIdx.x, tid = threadIdx.x;
  __shared__ unsigned char selb[300];
  __shared__ int cnts[5];
  __shared__ int anyf;
  __shared__ float pooledL[5 * 256];
  if (tid < 5) cnts[tid] = 0;
  if (tid == 0) anyf = 0;
  __syncthreads();
  for (int q = tid; q < 300; q += 256) {
    const float* lg = logits + ((size_t)b * 300 + q) * 5;
    unsigned bits = 0;
    #pragma unroll
    for (int p = 0; p < 5; ++p)
      if (lg[p] > THRL) { bits |= 1u << p; atomicAdd(&cnts[p], 1); }
    selb[q] = (unsigned char)bits;
    if (bits) atomicOr(&anyf, 1);
  }
  __syncthreads();
  const int d = tid;
  float a0 = 0, a1 = 0, a2 = 0, a3 = 0, a4 = 0;
  for (int q = 0; q < 300; ++q) {
    unsigned bb = selb[q];
    if (bb) {
      float h = hs[((size_t)b * 300 + q) * 256 + d];
      if (bb & 1)  a0 += h;
      if (bb & 2)  a1 += h;
      if (bb & 4)  a2 += h;
      if (bb & 8)  a3 += h;
      if (bb & 16) a4 += h;
    }
  }
  pooledL[0 * 256 + d] = cnts[0] > 0 ? a0 / (float)cnts[0] : 0.f;
  pooledL[1 * 256 + d] = cnts[1] > 0 ? a1 / (float)cnts[1] : 0.f;
  pooledL[2 * 256 + d] = cnts[2] > 0 ? a2 / (float)cnts[2] : 0.f;
  pooledL[3 * 256 + d] = cnts[3] > 0 ? a3 / (float)cnts[3] : 0.f;
  pooledL[4 * 256 + d] = cnts[4] > 0 ? a4 / (float)cnts[4] : 0.f;
  __syncthreads();
  const float asel = anyf ? 1.f : 0.f;
  for (int o = tid; o < 5 * 768; o += 256) {
    int p = o / 768, jj = o % 768;
    float acc = bq[jj];
    for (int d2 = 0; d2 < 256; ++d2)
      acc += pooledL[p * 256 + d2] * Wq[(size_t)d2 * 768 + jj];
    prompt[((size_t)b * 5 + p) * 768 + jj] = asel * acc;
  }
}

// ---------------- KE: query LN + combine with ctx mean + MLP head ----------------
__global__ __launch_bounds__(256) void kE_head(
    const float* __restrict__ ctxsum, const float* __restrict__ prompt,
    const float* __restrict__ lq,
    const float* __restrict__ lb0, const float* __restrict__ lb1,
    const float* __restrict__ lb2, const float* __restrict__ lb3,
    const float* __restrict__ qng, const float* __restrict__ qnb,
    const float* __restrict__ W1, const float* __restrict__ b1,
    const float* __restrict__ W2, const float* __restrict__ b2,
    float* __restrict__ out) {
  const int b = blockIdx.x, tid = threadIdx.x;
  __shared__ float mctx[768], yacc[768], rowb[768];
  __shared__ float red1[256], red2[256];
  for (int k = 0; k < 3; ++k) {
    int col = tid + 256 * k;
    float l = 1024.f * lb0[col] + 256.f * lb1[col] + 64.f * lb2[col] + 16.f * lb3[col];
    mctx[col] = (ctxsum[b * 768 + col] + l) * (1.f / 1360.f);
    yacc[col] = 0.f;
  }
  __syncthreads();
  for (int r = 0; r < 6; ++r) {
    float s1 = 0.f, s2 = 0.f;
    for (int k = 0; k < 3; ++k) {
      int col = tid + 256 * k;
      float xv = lq[(size_t)r * 768 + col];
      if (r > 0) xv += prompt[((size_t)b * 5 + (r - 1)) * 768 + col];
      rowb[col] = xv; s1 += xv; s2 += xv * xv;
    }
    red1[tid] = s1; red2[tid] = s2; __syncthreads();
    for (int off = 128; off; off >>= 1) {
      if (tid < off) { red1[tid] += red1[tid + off]; red2[tid] += red2[tid + off]; }
      __syncthreads();
    }
    float mean = red1[0] * (1.f / 768.f);
    float var  = red2[0] * (1.f / 768.f) - mean * mean;
    float rstd = rsqrtf(var + 1e-5f);
    float wgt  = (r == 0) ? 1.f : 0.2f;
    for (int k = 0; k < 3; ++k) {
      int col = tid + 256 * k;
      yacc[col] += wgt * (rowb[col] - mean) * rstd;
    }
    __syncthreads();
  }
  // y = qng*yacc + 2*qnb + 2*mctx  (last0 + mean(last1..5))
  for (int k = 0; k < 3; ++k) {
    int col = tid + 256 * k;
    rowb[col] = qng[col] * yacc[col] + 2.f * qnb[col] + 2.f * mctx[col];
  }
  __syncthreads();
  for (int k = 0; k < 3; ++k) {
    int j = tid + 256 * k;
    float acc = b1[j];
    for (int i = 0; i < 768; ++i) acc += rowb[i] * W1[(size_t)i * 768 + j];
    yacc[j] = fmaxf(acc, 0.f);
  }
  __syncthreads();
  for (int j = tid; j < 396; j += 256) {
    float acc = b2[j];
    for (int i = 0; i < 768; ++i) acc += yacc[i] * W2[(size_t)i * 396 + j];
    out[b * 396 + j] = acc;
  }
}

// ---------------- launch ----------------
extern "C" void kernel_launch(void* const* d_in, const int* in_sizes, int n_in,
                              void* d_out, int out_size, void* d_ws, size_t ws_size,
                              hipStream_t stream) {
  (void)in_sizes; (void)n_in; (void)out_size; (void)ws_size;
  const float* logits = (const float*)d_in[0];
  const float* hs     = (const float*)d_in[2];
  const float *F[4], *P[4], *CW[4], *CB[4], *LG[4], *LB[4];
  const unsigned char* M[4];
  for (int s = 0; s < 4; ++s) {
    F[s]  = (const float*)d_in[3 + 7 * s];
    P[s]  = (const float*)d_in[4 + 7 * s];
    M[s]  = (const unsigned char*)d_in[5 + 7 * s];
    CW[s] = (const float*)d_in[6 + 7 * s];
    CB[s] = (const float*)d_in[7 + 7 * s];
    LG[s] = (const float*)d_in[8 + 7 * s];
    LB[s] = (const float*)d_in[9 + 7 * s];
  }
  const float* lq  = (const float*)d_in[31];
  const float* Wq  = (const float*)d_in[32];
  const float* bq  = (const float*)d_in[33];
  const float* qng = (const float*)d_in[34];
  const float* qnb = (const float*)d_in[35];
  const float* W1  = (const float*)d_in[36];
  const float* b1  = (const float*)d_in[37];
  const float* W2  = (const float*)d_in[38];
  const float* b2  = (const float*)d_in[39];

  // workspace layout (bytes)
  char* ws = (char*)d_ws;
  unsigned short* pooled = (unsigned short*)(ws);                    // 8*1360*256*2 = 5,570,560
  unsigned short* wbf    = (unsigned short*)(ws + 5570560);          // 4*768*256*2  = 1,572,864
  float* mpinv           = (float*)(ws + 7143424);                   // 8*1360*4     = 43,520
  float* ctxsum          = (float*)(ws + 7186944);                   // 8*768*4      = 24,576
  float* prompt          = (float*)(ws + 7211520);                   // 8*5*768*4    = 122,880

  hipLaunchKernelGGL(kA_pool, dim3(NB * NTILES), dim3(256), 0, stream,
                     F[0], F[1], F[2], F[3], P[0], P[1], P[2], P[3],
                     M[0], M[1], M[2], M[3], pooled, mpinv);
  hipLaunchKernelGGL(kB_prep, dim3(3072), dim3(256), 0, stream,
                     CW[0], CW[1], CW[2], CW[3], wbf, ctxsum);
  hipLaunchKernelGGL(kC_gemm_ln, dim3(NB * NTILES), dim3(256), 0, stream,
                     pooled, wbf, mpinv,
                     CB[0], CB[1], CB[2], CB[3], LG[0], LG[1], LG[2], LG[3],
                     ctxsum);
  hipLaunchKernelGGL(kD_prompt, dim3(NB), dim3(256), 0, stream,
                     logits, hs, Wq, bq, prompt);
  hipLaunchKernelGGL(kE_head, dim3(NB), dim3(256), 0, stream,
                     ctxsum, prompt, lq, LB[0], LB[1], LB[2], LB[3],
                     qng, qnb, W1, b1, W2, b2, (float*)d_out);
}
